// SelfAttention_43147241455729
// MI455X (gfx1250) — compile-verified, matrix operand order checked by reading, not run on hardware
//
#include <hip/hip_runtime.h>

#define B_  2
#define S_  2048
#define D_  2048
#define H_  16
#define DH_ 128

typedef _Float16 v16h __attribute__((ext_vector_type(16)));
typedef _Float16 v8h  __attribute__((ext_vector_type(8)));
typedef float    v8f  __attribute__((ext_vector_type(8)));
typedef float    v4f  __attribute__((ext_vector_type(4)));
typedef unsigned int u32x4 __attribute__((ext_vector_type(4)));
typedef int      i32x4 __attribute__((ext_vector_type(4)));
typedef int      i32x8 __attribute__((ext_vector_type(8)));

#if defined(__has_builtin)
#  if __has_builtin(__builtin_amdgcn_tensor_load_to_lds) && \
      __has_builtin(__builtin_amdgcn_s_wait_tensorcnt)
#    define USE_TDM 1
#  endif
#endif
#ifndef USE_TDM
#  define USE_TDM 0
#endif

// D = A(16x32 f16) * B(32x16 f16) + C(16x16 f32)
__device__ __forceinline__ v8f wmma16(v16h a, v16h b, v8f c) {
  return __builtin_amdgcn_wmma_f32_16x16x32_f16(
      false, a, false, b, (short)0, c, false, false);
}

// 16-bit operand fragment: this lane holds K = kOff+0..7 (elems 0..7) and
// K = kOff+16..23 (elems 8..15); caller passes p already offset by kOff.
__device__ __forceinline__ v16h frag_from_f16(const _Float16* p) {
  v8h lo = *(const v8h*)(p);
  v8h hi = *(const v8h*)(p + 16);
  v16h r;
#pragma unroll
  for (int i = 0; i < 8; ++i) { r[i] = lo[i]; r[i + 8] = hi[i]; }
  return r;
}

// Butterfly reductions confined to each 16-lane half (xor masks < 16).
__device__ __forceinline__ float red_max16(float v) {
#pragma unroll
  for (int m = 1; m < 16; m <<= 1) v = fmaxf(v, __shfl_xor(v, m, 32));
  return v;
}
__device__ __forceinline__ float red_sum16(float v) {
#pragma unroll
  for (int m = 1; m < 16; m <<= 1) v += __shfl_xor(v, m, 32);
  return v;
}

#if USE_TDM
// Issue a TDM 2-D tile load (f16 elements) into LDS. D# packed per ISA §8.3/8.4:
// group0: count=1 | lds_addr | global_addr[56:0] | type=2
// group1: data_size=1(2B) | tensor_dim0/1 | tile_dim0/1 | tensor_dim0_stride
// Groups 2/3 zero (2-D tensor); 6-arg builtin form on this toolchain.
__device__ __forceinline__ void tdm_load_2d_f16(const void* lds_dst,
                                                const void* gsrc,
                                                unsigned tile_d0,   // elems/row
                                                unsigned tile_d1,   // rows
                                                unsigned long long stride0) {
  unsigned long long ga = (unsigned long long)(uintptr_t)gsrc;
  unsigned lds = (unsigned)(uintptr_t)lds_dst;   // low 32 bits = LDS offset
  u32x4 g0;
  g0[0] = 1u;                                          // count=1 (valid user D#)
  g0[1] = lds;                                         // lds_addr
  g0[2] = (unsigned)(ga & 0xFFFFFFFFu);                // global_addr[31:0]
  g0[3] = (unsigned)((ga >> 32) & 0x01FFFFFFu) | (2u << 30);  // [56:32]|type=2
  i32x8 g1;
  g1[0] = (int)(1u << 16);                             // data_size=1 -> 2 bytes
  g1[1] = (int)((tile_d0 & 0xFFFFu) << 16);            // tensor_dim0[15:0]
  g1[2] = (int)(((tile_d0 >> 16) & 0xFFFFu) |          // tensor_dim0[31:16]
                ((tile_d1 & 0xFFFFu) << 16));          // tensor_dim1[15:0]
  g1[3] = (int)(((tile_d1 >> 16) & 0xFFFFu) |          // tensor_dim1[31:16]
                ((tile_d0 & 0xFFFFu) << 16));          // tile_dim0
  g1[4] = (int)(tile_d1 & 0xFFFFu);                    // tile_dim1 (tile_dim2=0)
  g1[5] = (int)(unsigned)(stride0 & 0xFFFFFFFFull);    // dim0_stride[31:0]
  g1[6] = (int)(unsigned)((stride0 >> 32) & 0xFFFFull);// dim0_stride[47:32]
  g1[7] = 0;
  i32x4 z4 = {0, 0, 0, 0};
  i32x8 z8 = {0, 0, 0, 0, 0, 0, 0, 0};
  __builtin_amdgcn_tensor_load_to_lds(g0, g1, z4, z4, z8, 0);
}
#else
// Fallback: cooperative vectorized fill (all threads of the block).
__device__ __forceinline__ void lds_fill_tile(_Float16* dst, const _Float16* src,
                                              int rows, int rowlen,
                                              size_t src_stride) {
  const int vecs_per_row = rowlen >> 3;
  for (int idx = threadIdx.x; idx < rows * vecs_per_row; idx += blockDim.x) {
    int r = idx / vecs_per_row;
    int c = (idx - r * vecs_per_row) << 3;
    *(v8h*)(dst + r * rowlen + c) = *(const v8h*)(src + (size_t)r * src_stride + c);
  }
}
#endif

// ---------------------------------------------------------------------------
// Kernel 0: fp32 -> f16 pre-conversion (one-shot; removes in-loop converts).
// ---------------------------------------------------------------------------
__global__ __launch_bounds__(256)
void f32_to_f16_kernel(const float* __restrict__ src,
                       _Float16* __restrict__ dst, int n8) {
  int i = blockIdx.x * blockDim.x + threadIdx.x;
  if (i >= n8) return;
  v4f a = *(const v4f*)(src + (size_t)i * 8);
  v4f b = *(const v4f*)(src + (size_t)i * 8 + 4);
  v8h o;
#pragma unroll
  for (int j = 0; j < 4; ++j) { o[j] = (_Float16)a[j]; o[4 + j] = (_Float16)b[j]; }
  *(v8h*)(dst + (size_t)i * 8) = o;
}

// ---------------------------------------------------------------------------
// Kernel 1: fused QKV projection (WMMA f16, f32 accum) + RoPE.
// grid = (D/64, M/64, 3), 4 waves/block; wave -> 16x64 tile. The 64x32 weight
// tile is shared by all waves: double-buffered TDM stage into LDS.
// Outputs: Qh, Kh f16 (B,H,S,Dh); V transposed f16 (B,H,Dh,S).
// ---------------------------------------------------------------------------
__global__ __launch_bounds__(128)
void qkv_proj_rope_kernel(const _Float16* __restrict__ xh,
                          const _Float16* __restrict__ wqh,
                          const _Float16* __restrict__ wkh,
                          const _Float16* __restrict__ wvh,
                          const float* __restrict__ cosT,
                          const float* __restrict__ sinT,
                          _Float16* __restrict__ Qh,
                          _Float16* __restrict__ Kh,
                          _Float16* __restrict__ Vt) {
  __shared__ __align__(16) _Float16 wt[2][64][32];   // double-buffered B tile

  const int lane = threadIdx.x & 31;
  const int wave = threadIdx.x >> 5;
  const int half = lane >> 4;
  const int lr   = lane & 15;
  const int kOff = half * 8;

  const int mat = blockIdx.z;                        // 0=Q, 1=K, 2=V
  const _Float16* w = (mat == 0) ? wqh : (mat == 1) ? wkh : wvh;

  const int m0 = (blockIdx.y * 4 + wave) * 16;       // rows in B*S
  const int n0 = blockIdx.x * 64;                    // cols in D
  const _Float16* wbase = w + (size_t)n0 * D_;       // 64 rows, stride D

  const _Float16* ap_base = xh + (size_t)(m0 + lr) * D_ + kOff;

  v8f acc[4] = {v8f{}, v8f{}, v8f{}, v8f{}};

#if USE_TDM
  if (wave == 0)
    tdm_load_2d_f16(&wt[0][0][0], wbase, 32, 64, (unsigned long long)D_);
#endif

  for (int kb = 0; kb < D_; kb += 32) {
    const int cur = (kb >> 5) & 1;
#if USE_TDM
    if (wave == 0) __builtin_amdgcn_s_wait_tensorcnt(0);
    __syncthreads();                                  // tile 'cur' visible
    if (wave == 0 && kb + 32 < D_)                    // prefetch next tile
      tdm_load_2d_f16(&wt[cur ^ 1][0][0], wbase + kb + 32, 32, 64,
                      (unsigned long long)D_);
#else
    lds_fill_tile(&wt[cur][0][0], wbase + kb, 64, 32, (size_t)D_);
    __syncthreads();
#endif
    v16h a = frag_from_f16(ap_base + kb);
#pragma unroll
    for (int t = 0; t < 4; ++t)
      acc[t] = wmma16(a, frag_from_f16(&wt[cur][t * 16 + lr][kOff]), acc[t]);
    __syncthreads();                                  // done reading 'cur'
  }

  // Epilogue: C-layout elem r -> row m0 + r + 8*half, col n0 + t*16 + lr.
#pragma unroll
  for (int t = 0; t < 4; ++t) {
    const int n   = n0 + t * 16 + lr;
    const int h   = n >> 7;
    const int dh  = n & (DH_ - 1);
    const int odd = n & 1;
#pragma unroll
    for (int r = 0; r < 8; ++r) {
      const int mrow = m0 + r + 8 * half;
      const int b = mrow >> 11;                       // S_ = 2048
      const int s = mrow & (S_ - 1);
      float v = acc[t][r];
      if (mat < 2) {
        // RoPE interleaved: partner column n^1 lives in lane^1 (same elem).
        float p   = __shfl_xor(v, 1, 32);
        float rot = odd ? p : -p;
        v = v * cosT[(size_t)s * DH_ + dh] + rot * sinT[(size_t)s * DH_ + dh];
      }
      if (mat == 0)
        Qh[(((size_t)b * H_ + h) * S_ + s) * DH_ + dh] = (_Float16)v;
      else if (mat == 1)
        Kh[(((size_t)b * H_ + h) * S_ + s) * DH_ + dh] = (_Float16)v;
      else
        Vt[(((size_t)b * H_ + h) * DH_ + dh) * S_ + s] = (_Float16)v;
    }
  }
}

// ---------------------------------------------------------------------------
// Kernel 2: causal flash attention + final softmax over Dh.
// grid = (S/64, H, B), 4 waves/block; wave -> 16 query rows. K/V tiles for the
// 32-key chunk are TDM-staged into LDS and shared by all 4 waves.
// ---------------------------------------------------------------------------
__global__ __launch_bounds__(128)
void flash_attn_kernel(const _Float16* __restrict__ Qh,
                       const _Float16* __restrict__ Kh,
                       const _Float16* __restrict__ Vt,
                       float* __restrict__ out) {
  __shared__ __align__(16) _Float16 kt[32][128];       // keys   (8 KB)
  __shared__ __align__(16) _Float16 vt[128][32];       // V^T    (8 KB)
  __shared__ __align__(16) _Float16 pbuf[4][16][32];   // P transpose staging

  const int lane = threadIdx.x & 31;
  const int wave = threadIdx.x >> 5;
  const int half = lane >> 4;
  const int lr   = lane & 15;
  const int kOff = half * 8;

  const int b      = blockIdx.z;
  const int h      = blockIdx.y;
  const int q0_blk = blockIdx.x * 64;
  const int q0     = q0_blk + wave * 16;

  const _Float16* Qbh = Qh + ((size_t)b * H_ + h) * (size_t)S_ * DH_;
  const _Float16* Kbh = Kh + ((size_t)b * H_ + h) * (size_t)S_ * DH_;
  const _Float16* Vbh = Vt + ((size_t)b * H_ + h) * (size_t)DH_ * S_;

  // Q fragments: 4 chunks of 32 along Dh (A-operand, row q0+lr).
  v16h qf[4];
#pragma unroll
  for (int c = 0; c < 4; ++c)
    qf[c] = frag_from_f16(Qbh + (size_t)(q0 + lr) * DH_ + c * 32 + kOff);

  float m_run[8], l_run[8];
  v8f oacc[8];
#pragma unroll
  for (int r = 0; r < 8; ++r) { m_run[r] = -1e30f; l_run[r] = 0.0f; }
#pragma unroll
  for (int t = 0; t < 8; ++t) oacc[t] = v8f{};

  const float scale = 0.08838834764831845f;            // 1/sqrt(128)

  for (int k0 = 0; k0 <= q0_blk + 48; k0 += 32) {
#if USE_TDM
    if (wave == 0) {
      tdm_load_2d_f16(&kt[0][0], Kbh + (size_t)k0 * DH_, 128, 32,
                      (unsigned long long)DH_);
      tdm_load_2d_f16(&vt[0][0], Vbh + k0, 32, 128, (unsigned long long)S_);
      __builtin_amdgcn_s_wait_tensorcnt(0);
    }
#else
    lds_fill_tile(&kt[0][0], Kbh + (size_t)k0 * DH_, 32, 128, (size_t)DH_);
    lds_fill_tile(&vt[0][0], Vbh + k0, 128, 32, (size_t)S_);
#endif
    __syncthreads();

    if (k0 <= q0) {
      // ---- scores: two 16x16 tiles covering keys [k0, k0+32) ----
      v8f s0 = {}, s1 = {};
#pragma unroll
      for (int c = 0; c < 4; ++c) {
        s0 = wmma16(qf[c], frag_from_f16(&kt[lr][c * 32 + kOff]), s0);
        s1 = wmma16(qf[c], frag_from_f16(&kt[16 + lr][c * 32 + kOff]), s1);
      }

      // ---- causal mask + online softmax (row r -> q0 + r + 8*half) ----
#pragma unroll
      for (int r = 0; r < 8; ++r) {
        const int qrow = r + 8 * half;
        float e0 = s0[r] * scale;
        float e1 = s1[r] * scale;
        if (k0 + lr      > q0 + qrow) e0 = -1e30f;
        if (k0 + 16 + lr > q0 + qrow) e1 = -1e30f;
        float mx   = red_max16(fmaxf(e0, e1));
        float mnew = fmaxf(m_run[r], mx);
        float corr = __expf(m_run[r] - mnew);
        float p0   = __expf(e0 - mnew);
        float p1   = __expf(e1 - mnew);
        float rsum = red_sum16(p0 + p1);
        l_run[r] = l_run[r] * corr + rsum;
        m_run[r] = mnew;
#pragma unroll
        for (int t = 0; t < 8; ++t) oacc[t][r] *= corr;
        pbuf[wave][qrow][lr]      = (_Float16)p0;   // stage P (C-layout)
        pbuf[wave][qrow][16 + lr] = (_Float16)p1;
      }

      // reload P as 16x32 A-operand (same-wave DS ordering guarantees data)
      v16h pf = frag_from_f16(&pbuf[wave][lr][kOff]);

      // ---- out += P @ V : 8 N-tiles of 16 covering Dh=128 ----
#pragma unroll
      for (int t = 0; t < 8; ++t)
        oacc[t] = wmma16(pf, frag_from_f16(&vt[t * 16 + lr][kOff]), oacc[t]);
    }
    __syncthreads();                                  // before tile overwrite
  }

  // ---- normalize, second softmax over Dh, store fp32 (B,S,H*Dh) ----
#pragma unroll
  for (int r = 0; r < 8; ++r) {
    const float inv = 1.0f / l_run[r];
#pragma unroll
    for (int t = 0; t < 8; ++t) oacc[t][r] *= inv;

    float mx = -1e30f;
#pragma unroll
    for (int t = 0; t < 8; ++t) mx = fmaxf(mx, oacc[t][r]);
    mx = red_max16(mx);
    float sum = 0.0f;
#pragma unroll
    for (int t = 0; t < 8; ++t) {
      float e = __expf(oacc[t][r] - mx);
      oacc[t][r] = e;
      sum += e;
    }
    sum = red_sum16(sum);
    const float is = 1.0f / sum;

    const int s = q0 + r + 8 * half;
    float* orow = out + ((size_t)b * S_ + s) * D_ + h * DH_;
#pragma unroll
    for (int t = 0; t < 8; ++t) orow[t * 16 + lr] = oacc[t][r] * is;
  }
}

// ---------------------------------------------------------------------------
extern "C" void kernel_launch(void* const* d_in, const int* in_sizes, int n_in,
                              void* d_out, int out_size, void* d_ws, size_t ws_size,
                              hipStream_t stream) {
  (void)in_sizes; (void)n_in; (void)out_size; (void)ws_size;
  const float* x    = (const float*)d_in[0];
  const float* wq   = (const float*)d_in[1];
  const float* wk   = (const float*)d_in[2];
  const float* wv   = (const float*)d_in[3];
  const float* cosT = (const float*)d_in[4];
  const float* sinT = (const float*)d_in[5];
  float* out = (float*)d_out;

  const size_t NX = (size_t)B_ * S_ * D_;     // 8,388,608
  const size_t NW = (size_t)D_ * D_;          // 4,194,304
  const size_t NQ = (size_t)B_ * H_ * S_ * DH_;

  _Float16* xh  = (_Float16*)d_ws;
  _Float16* wqh = xh  + NX;
  _Float16* wkh = wqh + NW;
  _Float16* wvh = wkh + NW;
  _Float16* Qh  = wvh + NW;
  _Float16* Kh  = Qh  + NQ;
  _Float16* Vt  = Kh  + NQ;

  // 0) one-shot fp32 -> f16 conversion of activations + weights
  f32_to_f16_kernel<<<(int)(NX / 8 / 256), 256, 0, stream>>>(x, xh, (int)(NX / 8));
  f32_to_f16_kernel<<<(int)(NW / 8 / 256), 256, 0, stream>>>(wq, wqh, (int)(NW / 8));
  f32_to_f16_kernel<<<(int)(NW / 8 / 256), 256, 0, stream>>>(wk, wkh, (int)(NW / 8));
  f32_to_f16_kernel<<<(int)(NW / 8 / 256), 256, 0, stream>>>(wv, wvh, (int)(NW / 8));

  // 1) QKV + RoPE: 32 N-tiles x 64 M-blocks x {q,k,v}
  dim3 gp(D_ / 64, (B_ * S_) / 64, 3);
  qkv_proj_rope_kernel<<<gp, 128, 0, stream>>>(xh, wqh, wkh, wvh, cosT, sinT,
                                               Qh, Kh, Vt);

  // 2) Attention: 32 q-blocks x 16 heads x 2 batches
  dim3 ga(S_ / 64, H_, B_);
  flash_attn_kernel<<<ga, 128, 0, stream>>>(Qh, Kh, Vt, out);
}
